// HaloWindowAttention_69818988363877
// MI455X (gfx1250) — matrix operand order repaired and checked by reading, not compiled, not run on hardware
//
#include <hip/hip_runtime.h>

// ---------------------------------------------------------------------------
// HaloWindowAttention on MI455X (gfx1250), wave32 + WMMA bf16 + TDM.
// Pipeline: prep(bf16 pack/transpose) -> QKV GEMM -> flash attention -> proj.
// ---------------------------------------------------------------------------

typedef __bf16 v16bf __attribute__((ext_vector_type(16)));
typedef float  v8f   __attribute__((ext_vector_type(8)));
typedef unsigned int u32x4 __attribute__((ext_vector_type(4)));
typedef int          i32x4 __attribute__((ext_vector_type(4)));
typedef int          i32x8 __attribute__((ext_vector_type(8)));

#if defined(__has_builtin)
#if __has_builtin(__builtin_amdgcn_tensor_load_to_lds)
#define HAVE_TDM 1
#endif
#endif

union Frag {
  v16bf v;
  unsigned int u[8];
};

__device__ __forceinline__ unsigned short f2bf(float f) {
  unsigned int u = __float_as_uint(f);
  u += 0x7FFFu + ((u >> 16) & 1u);        // round-to-nearest-even
  return (unsigned short)(u >> 16);
}

__device__ __forceinline__ v8f wmma_bf16(const Frag& a, const Frag& b, v8f c) {
  return __builtin_amdgcn_wmma_f32_16x16x32_bf16(
      /*neg_a=*/false, a.v, /*neg_b=*/false, b.v,
      /*c_mod=*/(short)0, c, /*reuse_a=*/false, /*reuse_b=*/false);
}

// load 16 halves as two 16B chunks separated by gap_halves
__device__ __forceinline__ void load_frag(Frag& f, const unsigned short* p, int gap_halves) {
  *(uint4*)&f.u[0] = *(const uint4*)(p);
  *(uint4*)&f.u[4] = *(const uint4*)(p + gap_halves);
}

#ifdef HAVE_TDM
// ---------------------------------------------------------------------------
// Tensor Data Mover: DMA a (tile_w x tile_h [x tile_d]) bf16 tile from global
// into LDS with trailing-row padding (pad_interval/pad_amount per ISA 8.4).
// D# bitfields per CDNA5 ISA.
// ---------------------------------------------------------------------------
__device__ __forceinline__ void tdm_load_tile(
    unsigned lds_off, const unsigned short* gptr,
    unsigned tensor_w, unsigned tensor_h, unsigned tensor_d,
    unsigned tile_w, unsigned tile_h, unsigned tile_d,
    unsigned row_stride, unsigned d2_stride,
    unsigned pad_int, unsigned pad_amt) {
  unsigned long long ga = (unsigned long long)gptr;
  u32x4 g0;
  g0[0] = 1u;                                          // count=1 (valid), user mode
  g0[1] = lds_off;                                     // lds_addr @63:32
  g0[2] = (unsigned)(ga & 0xFFFFFFFFu);                // global_addr @95:64
  g0[3] = (unsigned)((ga >> 32) & 0x01FFFFFFu)         // global_addr hi @120:96
          | (2u << 30);                                // type=2 ("image") @127:126
  i32x8 g1;
  g1[0] = (int)((1u << 16)                             // data_size=1 -> 2 bytes
                | (1u << 20)                           // pad_enable
                | (pad_int << 22)                      // pad_interval
                | (pad_amt << 25));                    // pad_amount
  g1[1] = (int)((tensor_w & 0xFFFFu) << 16);           // tensor_dim0 lo16 @63:48
  g1[2] = (int)(((tensor_w >> 16) & 0xFFFFu)           // tensor_dim0 hi16 @79:64
                | ((tensor_h & 0xFFFFu) << 16));       // tensor_dim1 lo16 @95:80
  g1[3] = (int)(((tensor_h >> 16) & 0xFFFFu)           // tensor_dim1 hi16 @111:96
                | ((tile_w & 0xFFFFu) << 16));         // tile_dim0 @127:112
  g1[4] = (int)((tile_h & 0xFFFFu)                     // tile_dim1 @143:128
                | ((tile_d & 0xFFFFu) << 16));         // tile_dim2 @159:144
  g1[5] = (int)row_stride;                             // tensor_dim0_stride lo32 @191:160
  g1[6] = (int)((d2_stride & 0xFFFFu) << 16);          // dim0_stride hi16=0; dim1_stride lo16 @223:208
  g1[7] = (int)(d2_stride >> 16);                      // tensor_dim1_stride hi32 @255:224
  i32x4 g2;
  g2[0] = (int)tensor_d;                               // tensor_dim2 @31:0
  g2[1] = 0; g2[2] = 0; g2[3] = 0;                     // tensor_dim3 / dim2_stride / tile_dim3
  i32x4 g3 = {0, 0, 0, 0};
#if __clang_major__ >= 23
  i32x8 g4 = {0, 0, 0, 0, 0, 0, 0, 0};
  __builtin_amdgcn_tensor_load_to_lds(g0, g1, g2, g3, g4, 0);
#else
  __builtin_amdgcn_tensor_load_to_lds(g0, g1, g2, g3, 0);
#endif
}
#endif  // HAVE_TDM

// ---------------------------------------------------------------------------
// Kernel 1a: build windowed reflect-padded input, fp32 -> bf16
// xw[b][t][c], t = tr*32+tc over padded 32x32 window
// ---------------------------------------------------------------------------
__global__ void prep_xw_kernel(unsigned short* __restrict__ xw,
                               const float* __restrict__ x) {
  const int total = 16 * 1024 * 1024;
  int stride = gridDim.x * blockDim.x;
  for (int idx = blockIdx.x * blockDim.x + threadIdx.x; idx < total; idx += stride) {
    int c = idx & 1023;
    int t = (idx >> 10) & 1023;
    int b = idx >> 20;
    int tr = t >> 5, tc = t & 31;
    int ry = tr - 8; ry = ry < 0 ? -ry : (ry > 15 ? 30 - ry : ry);
    int rx = tc - 8; rx = rx < 0 ? -rx : (rx > 15 ? 30 - rx : rx);
    float val = x[(size_t)((b << 10) + c) * 256 + (ry << 4) + rx];
    xw[idx] = f2bf(val);
  }
}

// ---------------------------------------------------------------------------
// Kernel 1b: transpose fp32 weight [K][N] -> bf16 [N][K]
// ---------------------------------------------------------------------------
__global__ void transpose_bf16_kernel(unsigned short* __restrict__ dst,
                                      const float* __restrict__ src,
                                      int N, int K) {
  int total = N * K;
  int stride = gridDim.x * blockDim.x;
  for (int idx = blockIdx.x * blockDim.x + threadIdx.x; idx < total; idx += stride) {
    int k = idx % K;
    int n = idx / K;
    dst[idx] = f2bf(src[(size_t)k * N + n]);
  }
}

// ---------------------------------------------------------------------------
// Shared GEMM tile machinery: 128x128 C tile, BK=64, LDS row stride 80 halves
// (TDM pads 64-half rows with 16 halves: pad_interval=4 (32 DW), pad_amount=7
// (8 DW)). 8 waves in a 2x4 grid, each wave 64x32 -> 8 wmma accums, 2 K-substeps.
// ---------------------------------------------------------------------------
#define LSTRIDE 80

__device__ __forceinline__ void gemm_stage_compute(
    const unsigned short* As, const unsigned short* Bs,
    int mw, int nw, int li, int hi, v8f acc[4][2]) {
#pragma unroll
  for (int s = 0; s < 2; ++s) {
    Frag afr[4], bfr[2];
#pragma unroll
    for (int i = 0; i < 4; ++i) {
      int rr = mw * 64 + i * 16 + li;
      load_frag(afr[i], &As[rr * LSTRIDE + s * 32 + hi * 16], 8);
    }
#pragma unroll
    for (int j = 0; j < 2; ++j) {
      int cc = nw * 32 + j * 16 + li;
      load_frag(bfr[j], &Bs[cc * LSTRIDE + s * 32 + hi * 16], 8);
    }
#pragma unroll
    for (int i = 0; i < 4; ++i)
#pragma unroll
      for (int j = 0; j < 2; ++j)
        acc[i][j] = wmma_bf16(afr[i], bfr[j], acc[i][j]);
  }
}

#ifndef HAVE_TDM
// fallback: cooperative 256-thread copy of a 128x64-half tile into LDS
__device__ __forceinline__ void manual_stage(unsigned short* dstLds,
                                             const unsigned short* g,
                                             size_t gstride, int tid) {
  int lrow = tid >> 1;
  int lhalf = (tid & 1) * 32;
  const unsigned short* src = g + (size_t)lrow * gstride + lhalf;
  unsigned short* dst = &dstLds[lrow * LSTRIDE + lhalf];
  *(uint4*)(dst)      = *(const uint4*)(src);
  *(uint4*)(dst + 8)  = *(const uint4*)(src + 8);
  *(uint4*)(dst + 16) = *(const uint4*)(src + 16);
  *(uint4*)(dst + 24) = *(const uint4*)(src + 24);
  __builtin_prefetch(src + 64, 0, 0);
}
#endif

// ---------------------------------------------------------------------------
// Kernel 2: QKV GEMM  C[b] = xw[b](1024x1024) @ w_qkv(1024x3072) + bias
// Scatter epilogue into q[b,h,S,64] / k[b,h,S,64] / vT[b,h,64,S] (bf16).
// ---------------------------------------------------------------------------
__global__ __launch_bounds__(256) void qkv_gemm_kernel(
    const unsigned short* __restrict__ xw, const unsigned short* __restrict__ wT,
    const float* __restrict__ bias,
    unsigned short* __restrict__ q, unsigned short* __restrict__ k,
    unsigned short* __restrict__ vT) {
  __shared__ __align__(16) unsigned short As[128 * LSTRIDE];
  __shared__ __align__(16) unsigned short Bs[128 * LSTRIDE];

  int tid = threadIdx.x;
  int blk = blockIdx.x;
  int b  = blk / 192;
  int r  = blk % 192;
  int m0 = (r / 24) * 128;
  int n0 = (r % 24) * 128;

  const unsigned short* A = xw + (size_t)b * 1024 * 1024;

  int wave = tid >> 5, lane = tid & 31;
  int li = lane & 15, hi = lane >> 4;
  int mw = wave >> 2, nw = wave & 3;

  v8f acc[4][2];
#pragma unroll
  for (int i = 0; i < 4; ++i)
#pragma unroll
    for (int j = 0; j < 2; ++j) acc[i][j] = {};

#ifdef HAVE_TDM
  unsigned asOff = (unsigned)(size_t)(void*)As;
  unsigned bsOff = (unsigned)(size_t)(void*)Bs;
#endif

  for (int kt = 0; kt < 16; ++kt) {
    int k0 = kt * 64;
    __syncthreads();
#ifdef HAVE_TDM
    if (wave == 0) {
      tdm_load_tile(asOff, A + (size_t)m0 * 1024 + k0,
                    1024, 1024, 0, 64, 128, 0, 1024, 0, 4, 7);
      tdm_load_tile(bsOff, wT + (size_t)n0 * 1024 + k0,
                    1024, 3072, 0, 64, 128, 0, 1024, 0, 4, 7);
      __builtin_amdgcn_s_wait_tensorcnt(0);
    }
#else
    manual_stage(As, A + (size_t)m0 * 1024 + k0, 1024, tid);
    manual_stage(Bs, wT + (size_t)n0 * 1024 + k0, 1024, tid);
#endif
    __syncthreads();
    gemm_stage_compute(As, Bs, mw, nw, li, hi, acc);
  }

  // Epilogue: C-layout element (M = v + 8*hi, N = li)
#pragma unroll
  for (int i = 0; i < 4; ++i) {
#pragma unroll
    for (int j = 0; j < 2; ++j) {
#pragma unroll
      for (int v = 0; v < 8; ++v) {
        int t = m0 + mw * 64 + i * 16 + v + hi * 8;   // token
        int n = n0 + nw * 32 + j * 16 + li;           // qkv channel
        float val = acc[i][j][v] + bias[n];
        unsigned short bv = f2bf(val);
        int three = n >> 10;
        int hh = (n >> 6) & 15;
        int d  = n & 63;
        size_t base = (size_t)(b * 16 + hh);
        if (three == 0)      q[(base * 1024 + t) * 64 + d] = bv;
        else if (three == 1) k[(base * 1024 + t) * 64 + d] = bv;
        else                 vT[(base * 64 + d) * 1024 + t] = bv;
      }
    }
  }
}

// ---------------------------------------------------------------------------
// Kernel 3: flash attention per (b, h, 16-query tile). One wave per q-tile.
// Only q-tiles 16..47 (spatial rows 8..23) matter after the final crop.
// ---------------------------------------------------------------------------
__device__ __forceinline__ bool is_center(int t) {
  int r = t >> 5, c = t & 31;
  return (r >= 8) & (r < 24) & (c >= 8) & (c < 24);
}

__global__ __launch_bounds__(256) void attn_kernel(
    const unsigned short* __restrict__ qbuf, const unsigned short* __restrict__ kbuf,
    const unsigned short* __restrict__ vTbuf, unsigned short* __restrict__ aout) {
  __shared__ __align__(16) unsigned short ptile[8][16][48];  // per-wave P staging

  const float kScale = 0.125f;  // (1024/16)^-0.5
  int tid = threadIdx.x;
  int wave = tid >> 5, lane = tid & 31;
  int li = lane & 15, hi = lane >> 4;

  int b  = blockIdx.x >> 6;
  int h  = (blockIdx.x >> 2) & 15;
  int qg = blockIdx.x & 3;
  int qtile = 16 + qg * 8 + wave;     // 16..47 -> center band rows
  int qbase = qtile * 16;

  const unsigned short* qbh = qbuf  + (size_t)(b * 16 + h) * 1024 * 64;
  const unsigned short* kbh = kbuf  + (size_t)(b * 16 + h) * 1024 * 64;
  const unsigned short* vbh = vTbuf + (size_t)(b * 16 + h) * 64 * 1024;

  // q A-fragments for K=0..31 and K=32..63
  Frag qa0, qa1;
  {
    const unsigned short* p = qbh + (size_t)(qbase + li) * 64 + hi * 16;
    load_frag(qa0, p, 8);
    load_frag(qa1, p + 32, 8);
  }

  float m[8], l[8];
  v8f o[4];
#pragma unroll
  for (int v = 0; v < 8; ++v) { m[v] = -1e30f; l[v] = 0.f; }
#pragma unroll
  for (int t4 = 0; t4 < 4; ++t4) o[t4] = {};
  bool qc[8];
#pragma unroll
  for (int v = 0; v < 8; ++v) qc[v] = is_center(qbase + v + hi * 8);

  for (int kt = 0; kt < 32; ++kt) {
    int kbase = kt * 32;

    // k B-fragments: two 16-key column tiles, K(=hd) 0..31 / 32..63
    Frag k00, k01, k10, k11;
    {
      const unsigned short* p0 = kbh + (size_t)(kbase + li) * 64 + hi * 16;
      load_frag(k00, p0, 8);
      load_frag(k01, p0 + 32, 8);
      const unsigned short* p1 = p0 + 16 * 64;
      load_frag(k10, p1, 8);
      load_frag(k11, p1 + 32, 8);
      if (kt + 1 < 32) {
        __builtin_prefetch(p0 + 32 * 64, 0, 0);   // next key tile
        __builtin_prefetch(p1 + 32 * 64, 0, 0);
      }
    }

    v8f s0 = {}, s1 = {};
    s0 = wmma_bf16(qa0, k00, s0);
    s0 = wmma_bf16(qa1, k01, s0);
    s1 = wmma_bf16(qa0, k10, s1);
    s1 = wmma_bf16(qa1, k11, s1);

    bool kc0 = is_center(kbase + li);
    bool kc1 = is_center(kbase + 16 + li);

    float p0e[8], p1e[8];
#pragma unroll
    for (int v = 0; v < 8; ++v) {
      float a  = s0[v] * kScale + ((qc[v] && !kc0) ? -0.1f : 0.f);
      float bb = s1[v] * kScale + ((qc[v] && !kc1) ? -0.1f : 0.f);
      float mx = fmaxf(a, bb);
      mx = fmaxf(mx, __shfl_xor(mx, 1, 16));
      mx = fmaxf(mx, __shfl_xor(mx, 2, 16));
      mx = fmaxf(mx, __shfl_xor(mx, 4, 16));
      mx = fmaxf(mx, __shfl_xor(mx, 8, 16));
      float mn = fmaxf(m[v], mx);
      float alpha = __expf(m[v] - mn);
      float ea = __expf(a - mn);
      float eb = __expf(bb - mn);
      float rs = ea + eb;
      rs += __shfl_xor(rs, 1, 16);
      rs += __shfl_xor(rs, 2, 16);
      rs += __shfl_xor(rs, 4, 16);
      rs += __shfl_xor(rs, 8, 16);
      l[v] = l[v] * alpha + rs;
      m[v] = mn;
      o[0][v] *= alpha; o[1][v] *= alpha; o[2][v] *= alpha; o[3][v] *= alpha;
      p0e[v] = ea; p1e[v] = eb;
    }

    // C-layout -> A-layout via per-wave LDS tile (16 rows x 32 keys, bf16)
#pragma unroll
    for (int v = 0; v < 8; ++v) {
      int r = v + hi * 8;
      ptile[wave][r][li]      = f2bf(p0e[v]);
      ptile[wave][r][16 + li] = f2bf(p1e[v]);
    }
    Frag pa;
    load_frag(pa, &ptile[wave][li][hi * 16], 8);

    // P(16x32) x V(32x64): 4 N-tiles
#pragma unroll
    for (int t4 = 0; t4 < 4; ++t4) {
      Frag vb;
      const unsigned short* vp = vbh + (size_t)(t4 * 16 + li) * 1024 + kbase + hi * 16;
      load_frag(vb, vp, 8);
      o[t4] = wmma_bf16(pa, vb, o[t4]);
    }
  }

  // normalize + store attn_out[b][token][h*64 + d] (bf16)
#pragma unroll
  for (int t4 = 0; t4 < 4; ++t4) {
#pragma unroll
    for (int v = 0; v < 8; ++v) {
      int token = qbase + v + hi * 8;
      float val = o[t4][v] / l[v];
      aout[(size_t)(b * 1024 + token) * 1024 + h * 64 + t4 * 16 + li] = f2bf(val);
    }
  }
}

// ---------------------------------------------------------------------------
// Kernel 4: projection GEMM over the 256 center tokens per window.
// A rows = 8 chunks of 16 consecutive tokens -> 3D TDM tile (dim2 stride = 32
// rows). out[b][n][y][x] = attn_out[b][token(y,x)] . w_proj[:,n] + b_proj[n]
// ---------------------------------------------------------------------------
__global__ __launch_bounds__(256) void proj_gemm_kernel(
    const unsigned short* __restrict__ ain, const unsigned short* __restrict__ wT,
    const float* __restrict__ bias, float* __restrict__ out) {
  __shared__ __align__(16) unsigned short As[128 * LSTRIDE];
  __shared__ __align__(16) unsigned short Bs[128 * LSTRIDE];

  int tid = threadIdx.x;
  int blk = blockIdx.x;
  int b  = blk / 16;
  int r  = blk % 16;
  int m0 = (r / 8) * 128;   // center-token tile base (0 or 128)
  int n0 = (r % 8) * 128;

  const unsigned short* A = ain + (size_t)b * 1024 * 1024;
  int tok0 = ((m0 >> 4) + 8) * 32 + 8;   // first gathered token of this tile

  int wave = tid >> 5, lane = tid & 31;
  int li = lane & 15, hi = lane >> 4;
  int mw = wave >> 2, nw = wave & 3;

  v8f acc[4][2];
#pragma unroll
  for (int i = 0; i < 4; ++i)
#pragma unroll
    for (int j = 0; j < 2; ++j) acc[i][j] = {};

#ifdef HAVE_TDM
  unsigned asOff = (unsigned)(size_t)(void*)As;
  unsigned bsOff = (unsigned)(size_t)(void*)Bs;
#endif

  for (int kt = 0; kt < 16; ++kt) {
    int k0 = kt * 64;
    __syncthreads();
#ifdef HAVE_TDM
    if (wave == 0) {
      // A: 3D tile, 8 chunks x 16 rows x 64 halves, chunk stride = 32 rows
      tdm_load_tile(asOff, A + (size_t)tok0 * 1024 + k0,
                    1024, 16, 8, 64, 16, 8, 1024, 32 * 1024, 4, 7);
      tdm_load_tile(bsOff, wT + (size_t)n0 * 1024 + k0,
                    1024, 1024, 0, 64, 128, 0, 1024, 0, 4, 7);
      __builtin_amdgcn_s_wait_tensorcnt(0);
    }
#else
    {
      int lrow = tid >> 1;
      int ct = m0 + lrow;
      int tok = ((ct >> 4) + 8) * 32 + (ct & 15) + 8;
      int lhalf = (tid & 1) * 32;
      const unsigned short* src = A + (size_t)tok * 1024 + k0 + lhalf;
      unsigned short* dst = &As[lrow * LSTRIDE + lhalf];
      *(uint4*)(dst)      = *(const uint4*)(src);
      *(uint4*)(dst + 8)  = *(const uint4*)(src + 8);
      *(uint4*)(dst + 16) = *(const uint4*)(src + 16);
      *(uint4*)(dst + 24) = *(const uint4*)(src + 24);
    }
    manual_stage(Bs, wT + (size_t)n0 * 1024 + k0, 1024, tid);
#endif
    __syncthreads();
    gemm_stage_compute(As, Bs, mw, nw, li, hi, acc);
  }

#pragma unroll
  for (int i = 0; i < 4; ++i) {
#pragma unroll
    for (int j = 0; j < 2; ++j) {
#pragma unroll
      for (int v = 0; v < 8; ++v) {
        int ct = m0 + mw * 64 + i * 16 + v + hi * 8;
        int n  = n0 + nw * 32 + j * 16 + li;
        int y = ct >> 4, xx = ct & 15;
        float val = acc[i][j][v] + bias[n];
        out[((size_t)(b * 1024 + n) * 16 + y) * 16 + xx] = val;
      }
    }
  }
}

// ---------------------------------------------------------------------------
extern "C" void kernel_launch(void* const* d_in, const int* in_sizes, int n_in,
                              void* d_out, int out_size, void* d_ws, size_t ws_size,
                              hipStream_t stream) {
  const float* x      = (const float*)d_in[0];
  const float* w_qkv  = (const float*)d_in[1];
  const float* b_qkv  = (const float*)d_in[2];
  const float* w_proj = (const float*)d_in[3];
  const float* b_proj = (const float*)d_in[4];
  float* out = (float*)d_out;

  char* ws = (char*)d_ws;
  size_t off = 0;
  auto walloc = [&](size_t bytes) -> void* {
    void* p = ws + off;
    off += (bytes + 255) & ~(size_t)255;
    return p;
  };
  unsigned short* xw     = (unsigned short*)walloc((size_t)16 * 1024 * 1024 * 2);
  unsigned short* wqkvT  = (unsigned short*)walloc((size_t)3072 * 1024 * 2);
  unsigned short* wprojT = (unsigned short*)walloc((size_t)1024 * 1024 * 2);
  unsigned short* qbuf   = (unsigned short*)walloc((size_t)16 * 16 * 1024 * 64 * 2);
  unsigned short* kbuf   = (unsigned short*)walloc((size_t)16 * 16 * 1024 * 64 * 2);
  unsigned short* vTbuf  = (unsigned short*)walloc((size_t)16 * 16 * 1024 * 64 * 2);
  unsigned short* aout   = (unsigned short*)walloc((size_t)16 * 1024 * 1024 * 2);

  prep_xw_kernel<<<4096, 256, 0, stream>>>(xw, x);
  transpose_bf16_kernel<<<2048, 256, 0, stream>>>(wqkvT, w_qkv, 3072, 1024);
  transpose_bf16_kernel<<<1024, 256, 0, stream>>>(wprojT, w_proj, 1024, 1024);
  qkv_gemm_kernel<<<3072, 256, 0, stream>>>(xw, wqkvT, b_qkv, qbuf, kbuf, vTbuf);
  attn_kernel<<<1024, 256, 0, stream>>>(qbuf, kbuf, vTbuf, aout);
  proj_gemm_kernel<<<256, 256, 0, stream>>>(aout, wprojT, b_proj, out);
}
